// InterpretableMultiTaskCrystal_74079595921433
// MI455X (gfx1250) — matrix operand order repaired and checked by reading, not compiled, not run on hardware
//
#include <hip/hip_runtime.h>
#include <hip/hip_bf16.h>

// ---------------- constants ----------------
#define NN      20000
#define EE      320000
#define BB      64
#define HID     128
#define DEPTH   4
#define NFREQ   8
#define KD      259      // 2*HID + 3
#define KW      288      // KD padded to 9 chunks of 32
#define KL      296      // LDS row stride for edge tile (16B aligned, bank-staggered)
#define WC      144      // 128 msg cols + 1 att col + pad to 9 tiles of 16
#define TE      32       // edges per block
#define FKK     1024     // NFREQ*HID
#define FKL     1032     // LDS row stride for FK tile

typedef __attribute__((ext_vector_type(16))) __bf16 v16bf;
typedef __attribute__((ext_vector_type(8)))  __bf16 v8bf;
typedef __attribute__((ext_vector_type(8)))  float  v8f;
typedef int v4i_vs __attribute__((vector_size(16)));   // matches builtin param pointee

#if defined(__gfx1250__) && __has_builtin(__builtin_amdgcn_global_load_async_to_lds_b128)
#define HAVE_ASYNC_LDS 1
#else
#define HAVE_ASYNC_LDS 0
#endif

__device__ inline void wait_async0() {
#if defined(__gfx1250__)
#if __has_builtin(__builtin_amdgcn_s_wait_asynccnt)
  __builtin_amdgcn_s_wait_asynccnt(0);
#else
  asm volatile("s_wait_asynccnt 0x0" ::: "memory");
#endif
#endif
}

#if HAVE_ASYNC_LDS
// 16B global -> LDS via CDNA5 async DMA path (ASYNCcnt-tracked, no VGPR staging)
__device__ inline void async_copy16(const void* gsrc, void* ldst) {
  __builtin_amdgcn_global_load_async_to_lds_b128(
      (v4i_vs __attribute__((address_space(1)))*)gsrc,
      (v4i_vs __attribute__((address_space(3)))*)ldst,
      /*offset=*/0, /*cpol=*/0);
}
#endif

__device__ inline v8f wmma_bf16(v16bf a, v16bf b, v8f c) {
  // D = A(16x32 bf16) * B(32x16 bf16) + C(16x16 f32)
  return __builtin_amdgcn_wmma_f32_16x16x32_bf16(
      /*neg_a=*/false, a, /*neg_b=*/false, b,
      /*c_mod=*/(short)0, c, /*reuse_a=*/false, /*reuse_b=*/false);
}

__device__ inline v16bf mk_frag(v8bf lo, v8bf hi) {
  v16bf a;
#pragma unroll
  for (int p = 0; p < 8; ++p) { a[p] = lo[p]; a[p + 8] = hi[p]; }
  return a;
}

// ---------------- small utility kernels ----------------
__global__ void k_zero(float* __restrict__ p, int n) {
  int i = blockIdx.x * 256 + threadIdx.x;
  if (i < n) p[i] = 0.f;
}

__global__ void k_init_red(unsigned* redmax, float* redsum) {
  if (threadIdx.x == 0) { *redmax = 0u; *redsum = 0.f; }
}

// build extended transposed weights: out[l][col][k], col<128 -> lin_w,
// col==128 -> att_w (attention folded in as one extra GEMM column), else 0
__global__ void k_prep_lin(const float* __restrict__ lw, const float* __restrict__ aw,
                           __bf16* __restrict__ out) {
  int idx = blockIdx.x * 256 + threadIdx.x;
  if (idx >= DEPTH * WC * KW) return;
  int l = idx / (WC * KW);
  int r = idx - l * (WC * KW);
  int col = r / KW;
  int k   = r - col * KW;
  float v = 0.f;
  if (k < KD) {
    if (col < HID)       v = lw[(l * KD + k) * HID + col];
    else if (col == HID) v = aw[l * KD + k];
  }
  out[idx] = (__bf16)v;
}

// transpose fk_w (f32 [D][FKK][H]) -> bf16 [D][H][FKK]
__global__ void k_prep_fk(const float* __restrict__ fw, __bf16* __restrict__ out) {
  int idx = blockIdx.x * 256 + threadIdx.x;
  if (idx >= DEPTH * HID * FKK) return;
  int l = idx / (HID * FKK);
  int r = idx - l * (HID * FKK);
  int col = r / FKK;
  int k   = r - col * FKK;
  out[idx] = (__bf16)fw[(l * FKK + k) * HID + col];
}

// h = x @ node_w + node_b ; write f32 + bf16 mirrors
__global__ void k_embed(const float* __restrict__ x, const float* __restrict__ nw,
                        const float* __restrict__ nb, float* __restrict__ hA,
                        __bf16* __restrict__ hbf) {
  int idx = blockIdx.x * 256 + threadIdx.x;
  if (idx >= NN * HID) return;
  int n = idx >> 7, c = idx & 127;
  float s = nb[c];
#pragma unroll
  for (int j = 0; j < 4; ++j) s += x[n * 4 + j] * nw[j * HID + c];
  hA[idx] = s;
  hbf[idx] = (__bf16)s;
}

// ---------------- edge conv: async gather-to-LDS -> WMMA (msg + fused att col) -> scatter ----------------
__global__ __launch_bounds__(128)
void k_edge(const __bf16* __restrict__ hbf, const int* __restrict__ src,
            const int* __restrict__ dst, const float* __restrict__ eattr,
            const __bf16* __restrict__ WT,      // [WC][KW] bf16, layer slice
            const float* __restrict__ lin_b,    // [HID]
            const float* __restrict__ att_b,    // [1]
            float* __restrict__ hB) {
  const int tid = threadIdx.x;
  const int e0 = blockIdx.x * TE;

  __shared__ __bf16 sT[TE][KL];
  __shared__ float  sAl[TE];

  // gather: (row, 16-byte chunk) pairs; chunks 0..7 -> h[dst], 8..15 -> h[src]
  for (int idx = tid; idx < TE * 16; idx += 128) {
    int r = idx >> 4, c = idx & 15;
    int e = e0 + r;
    int ec = (e < EE) ? e : (EE - 1);
    const __bf16* gsrc;
    __bf16* ldst;
    if (c < 8) {
      gsrc = &hbf[dst[ec] * HID + c * 8];
      ldst = &sT[r][c * 8];
    } else {
      int cc = c - 8;
      gsrc = &hbf[src[ec] * HID + cc * 8];
      ldst = &sT[r][HID + cc * 8];
    }
#if HAVE_ASYNC_LDS
    async_copy16(gsrc, ldst);
#else
    *(v8bf*)ldst = *(const v8bf*)gsrc;
#endif
  }
  // tail: edge_attr + zero pad (cols 259..287)
  if (tid < TE) {
    int e = e0 + tid;
    if (e < EE) {
      sT[tid][256] = (__bf16)eattr[e * 3 + 0];
      sT[tid][257] = (__bf16)eattr[e * 3 + 1];
      sT[tid][258] = (__bf16)eattr[e * 3 + 2];
    } else {
      sT[tid][256] = sT[tid][257] = sT[tid][258] = (__bf16)0.f;
    }
    for (int k = KD; k < KW; ++k) sT[tid][k] = (__bf16)0.f;
  }
#if HAVE_ASYNC_LDS
  wait_async0();
#endif
  __syncthreads();

  // WMMA: wave w owns msg columns [w*32, w*32+32)
  const int waveu = __builtin_amdgcn_readfirstlane((int)threadIdx.x) >> 5; // scalar
  const int lane = tid & 31;
  const int g = lane >> 4, lm = lane & 15;
  const int n0 = waveu * 32;
  v8f c00 = {}, c01 = {}, c10 = {}, c11 = {};
  const __bf16* W0 = &WT[(n0 + lm) * KW];
  const __bf16* W1 = &WT[(n0 + 16 + lm) * KW];
  __builtin_prefetch((const void*)W0, 0, 1);
  __builtin_prefetch((const void*)W1, 0, 1);

#pragma unroll
  for (int kc = 0; kc < 9; ++kc) {      // branch-free main GEMM loop
    const int kb = kc * 32;
    v16bf a0 = mk_frag(*(const v8bf*)&sT[lm][kb + g * 8],
                       *(const v8bf*)&sT[lm][kb + 16 + g * 8]);
    v16bf a1 = mk_frag(*(const v8bf*)&sT[16 + lm][kb + g * 8],
                       *(const v8bf*)&sT[16 + lm][kb + 16 + g * 8]);
    v16bf b0 = mk_frag(*(const v8bf*)&W0[kb + g * 8], *(const v8bf*)&W0[kb + 16 + g * 8]);
    v16bf b1 = mk_frag(*(const v8bf*)&W1[kb + g * 8], *(const v8bf*)&W1[kb + 16 + g * 8]);
    c00 = wmma_bf16(a0, b0, c00);
    c10 = wmma_bf16(a1, b0, c10);
    c01 = wmma_bf16(a0, b1, c01);
    c11 = wmma_bf16(a1, b1, c11);
  }

  // attention column tile (col 128): wave 3 only, separate pass (scalar branch,
  // EXEC all-ones inside -> WMMA-legal)
  if (waveu == 3) {
    v8f c20 = {}, c21 = {};
    const __bf16* W2 = &WT[(HID + lm) * KW];
#pragma unroll
    for (int kc = 0; kc < 9; ++kc) {
      const int kb = kc * 32;
      v16bf a0 = mk_frag(*(const v8bf*)&sT[lm][kb + g * 8],
                         *(const v8bf*)&sT[lm][kb + 16 + g * 8]);
      v16bf a1 = mk_frag(*(const v8bf*)&sT[16 + lm][kb + g * 8],
                         *(const v8bf*)&sT[16 + lm][kb + 16 + g * 8]);
      v16bf b2 = mk_frag(*(const v8bf*)&W2[kb + g * 8], *(const v8bf*)&W2[kb + 16 + g * 8]);
      c20 = wmma_bf16(a0, b2, c20);
      c21 = wmma_bf16(a1, b2, c21);
    }
    float ab = att_b[0];
    if (lm == 0) {                       // lanes holding column 128
#pragma unroll
      for (int r = 0; r < 8; ++r) {
        int m = g * 8 + r;
        sAl[m]      = 1.f / (1.f + __expf(-(c20[r] + ab)));
        sAl[16 + m] = 1.f / (1.f + __expf(-(c21[r] + ab)));
      }
    }
  }
  __syncthreads();

  // msg = alpha * (acc + lin_b); scatter-add to h_conv[dst]
  const int col0 = n0 + lm, col1 = n0 + 16 + lm;
  const float lb0 = lin_b[col0], lb1 = lin_b[col1];
#pragma unroll
  for (int r = 0; r < 8; ++r) {
    int m = g * 8 + r;
    int eA = e0 + m, eB = e0 + 16 + m;
    if (eA < EE) {
      int dn = dst[eA];
      float al = sAl[m];
      atomicAdd(&hB[dn * HID + col0], al * (c00[r] + lb0));
      atomicAdd(&hB[dn * HID + col1], al * (c01[r] + lb1));
    }
    if (eB < EE) {
      int dn = dst[eB];
      float al = sAl[16 + m];
      atomicAdd(&hB[dn * HID + col0], al * (c10[r] + lb0));
      atomicAdd(&hB[dn * HID + col1], al * (c11[r] + lb1));
    }
  }
}

// ---------------- Fourier-Kolmogorov branch: (sin+cos) features -> WMMA GEMM ----------------
__global__ __launch_bounds__(128)
void k_fk(const float* __restrict__ hA, const __bf16* __restrict__ FWT, // [HID][FKK]
          const float* __restrict__ fk_b, float* __restrict__ hB) {
  const int tid = threadIdx.x;
  const int n0 = blockIdx.x * 16;

  __shared__ __bf16 sS[16][FKL];
  // packed v8bf feature build: chunk = 8 contiguous elements of k*128+i layout
  for (int idx = tid; idx < 16 * FKK / 8; idx += 128) {
    int m = idx >> 7;
    int off = (idx & 127) * 8;      // element offset in [0,1024), = k*128 + i
    int node = n0 + m;
    v8bf outv;
    if (node < NN) {
      int k = off >> 7, i = off & 127;
      float fs = 6.28318530718f * (float)(k + 1);
#pragma unroll
      for (int j = 0; j < 8; ++j) {
        float ang = fs * hA[node * HID + i + j];
        outv[j] = (__bf16)(__sinf(ang) + __cosf(ang));
      }
    } else {
#pragma unroll
      for (int j = 0; j < 8; ++j) outv[j] = (__bf16)0.f;
    }
    *(v8bf*)&sS[m][off] = outv;
  }
  __syncthreads();

  const int wave = tid >> 5, lane = tid & 31;
  const int g = lane >> 4, lm = lane & 15;
  const int nb = wave * 32;
  v8f c0 = {}, c1 = {};
  const __bf16* W0 = &FWT[(nb + lm) * FKK];
  const __bf16* W1 = &FWT[(nb + 16 + lm) * FKK];
  __builtin_prefetch((const void*)W0, 0, 1);
  __builtin_prefetch((const void*)W1, 0, 1);
#pragma unroll 4
  for (int kc = 0; kc < 32; ++kc) {
    const int kb = kc * 32;
    v16bf a = mk_frag(*(const v8bf*)&sS[lm][kb + g * 8],
                      *(const v8bf*)&sS[lm][kb + 16 + g * 8]);
    v16bf b0 = mk_frag(*(const v8bf*)&W0[kb + g * 8], *(const v8bf*)&W0[kb + 16 + g * 8]);
    v16bf b1 = mk_frag(*(const v8bf*)&W1[kb + g * 8], *(const v8bf*)&W1[kb + 16 + g * 8]);
    c0 = wmma_bf16(a, b0, c0);
    c1 = wmma_bf16(a, b1, c1);
  }

#pragma unroll
  for (int r = 0; r < 8; ++r) {
    int m = g * 8 + r;
    int node = n0 + m;
    if (node >= NN) continue;
    int col0 = nb + lm, col1 = nb + 16 + lm;
    hB[node * HID + col0] += c0[r] + fk_b[col0];   // exclusive (node,col) -> no atomic
    hB[node * HID + col1] += c1[r] + fk_b[col1];
  }
}

// ---------------- batch norm ----------------
__global__ __launch_bounds__(256)
void k_bn_stats(const float* __restrict__ hB, float* __restrict__ stats) {
  __shared__ float s1[256], s2[256];
  int tid = threadIdx.x;
  int col = tid & 127, half = tid >> 7;
  int base = blockIdx.x * 32;
  int rend = base + 32; if (rend > NN) rend = NN;
  float a = 0.f, b = 0.f;
  for (int r = base + half; r < rend; r += 2) {
    float v = hB[r * HID + col];
    a += v; b += v * v;
  }
  s1[tid] = a; s2[tid] = b;
  __syncthreads();
  if (tid < 128) {
    atomicAdd(&stats[col],       s1[tid] + s1[tid + 128]);
    atomicAdd(&stats[128 + col], s2[tid] + s2[tid + 128]);
  }
}

__global__ void k_bn_apply(const float* __restrict__ hB, const float* __restrict__ stats,
                           const float* __restrict__ g, const float* __restrict__ b,
                           float* __restrict__ hA, __bf16* __restrict__ hbf) {
  int idx = blockIdx.x * 256 + threadIdx.x;
  if (idx >= NN * HID) return;
  int c = idx & 127;
  const float invN = 1.f / (float)NN;
  float mu  = stats[c] * invN;
  float var = stats[128 + c] * invN - mu * mu;
  float y = (hB[idx] - mu) * rsqrtf(var + 1e-5f) * g[c] + b[c];
  y = fmaxf(y, 0.f);
  hA[idx] = y;
  hbf[idx] = (__bf16)y;
}

// ---------------- global softmax attention pool ----------------
__device__ inline unsigned fenc(float f) {
  unsigned u = __float_as_uint(f);
  return (u & 0x80000000u) ? ~u : (u | 0x80000000u);
}
__device__ inline float fdec(unsigned k) {
  return (k & 0x80000000u) ? __uint_as_float(k ^ 0x80000000u) : __uint_as_float(~k);
}

__global__ void k_logits(const float* __restrict__ hA, const float* __restrict__ pw,
                         const float* __restrict__ pb, float* __restrict__ logits,
                         unsigned* __restrict__ redmax) {
  int n = blockIdx.x * 256 + threadIdx.x;
  if (n >= NN) return;
  float s = pb[0];
#pragma unroll 8
  for (int c = 0; c < HID; ++c) s += hA[n * HID + c] * pw[c];
  logits[n] = s;
  atomicMax(redmax, fenc(s));
}

__global__ __launch_bounds__(256)
void k_expsum(float* __restrict__ logits, const unsigned* __restrict__ redmax,
              float* __restrict__ redsum) {
  __shared__ float sh[256];
  int n = blockIdx.x * 256 + threadIdx.x;
  float mx = fdec(*redmax);
  float v = 0.f;
  if (n < NN) { v = __expf(logits[n] - mx); logits[n] = v; }
  sh[threadIdx.x] = v;
  __syncthreads();
  for (int s = 128; s > 0; s >>= 1) {
    if (threadIdx.x < s) sh[threadIdx.x] += sh[threadIdx.x + s];
    __syncthreads();
  }
  if (threadIdx.x == 0) atomicAdd(redsum, sh[0]);
}

__global__ void k_pool(const float* __restrict__ hA, const float* __restrict__ logits,
                       const float* __restrict__ redsum, const int* __restrict__ batch,
                       float* __restrict__ pooled) {
  int idx = blockIdx.x * 256 + threadIdx.x;
  if (idx >= NN * HID) return;
  int n = idx >> 7, c = idx & 127;
  float a = logits[n] / redsum[0];
  atomicAdd(&pooled[batch[n] * HID + c], a * hA[idx]);
}

__global__ void k_comb(const float* __restrict__ pooled, const float* __restrict__ sg_table,
                       const int* __restrict__ sgid, float* __restrict__ comb) {
  int idx = blockIdx.x * 256 + threadIdx.x;
  if (idx >= BB * 2 * HID) return;
  int b = idx / (2 * HID), j = idx - b * (2 * HID);
  comb[idx] = (j < HID) ? pooled[b * HID + j]
                        : sg_table[sgid[b] * HID + (j - HID)];
}

// ---------------- multitask heads ----------------
__global__ __launch_bounds__(128)
void k_heads(const float* __restrict__ comb, const float* __restrict__ hw1,
             const float* __restrict__ hb1,
             const float* __restrict__ ew2, const float* __restrict__ eb2,
             const float* __restrict__ sw2, const float* __restrict__ sb2,
             const float* __restrict__ cw2, const float* __restrict__ cb2,
             const float* __restrict__ mw2, const float* __restrict__ mb2,
             float* __restrict__ out) {
  __shared__ float sc[2 * HID];
  __shared__ float sz[HID];
  int b = blockIdx.x;
  int tid = threadIdx.x;
  sc[tid] = comb[b * 2 * HID + tid];
  sc[tid + HID] = comb[b * 2 * HID + tid + HID];
  __syncthreads();

  const float* w2[4] = {ew2, sw2, cw2, mw2};
  const float* b2[4] = {eb2, sb2, cb2, mb2};
  const int od[4]  = {1, 3, 7, 3};
  const int off[4] = {0, BB * 1, BB * (1 + 3), BB * (1 + 3 + 7)};

  for (int i = 0; i < 4; ++i) {
    float acc = hb1[i * HID + tid];
    const float* W = &hw1[i * 2 * HID * HID];
#pragma unroll 8
    for (int k = 0; k < 2 * HID; ++k) acc += sc[k] * W[k * HID + tid];
    sz[tid] = fmaxf(acc, 0.f);
    __syncthreads();
    if (tid < od[i]) {
      float o = b2[i][tid];
      for (int j = 0; j < HID; ++j) o += sz[j] * w2[i][j * od[i] + tid];
      out[off[i] + b * od[i] + tid] = o;
    }
    __syncthreads();
  }
}

// ---------------- launcher ----------------
extern "C" void kernel_launch(void* const* d_in, const int* in_sizes, int n_in,
                              void* d_out, int out_size, void* d_ws, size_t ws_size,
                              hipStream_t stream) {
  const float* x         = (const float*)d_in[0];
  const int*   ei        = (const int*)  d_in[1];   // [2,E]: row0=src, row1=dst
  const float* eattr     = (const float*)d_in[2];
  const int*   batch     = (const int*)  d_in[3];
  const int*   sgid      = (const int*)  d_in[4];
  const float* node_w    = (const float*)d_in[5];
  const float* node_b    = (const float*)d_in[6];
  const float* sg_table  = (const float*)d_in[7];
  const float* lin_w     = (const float*)d_in[8];
  const float* lin_b     = (const float*)d_in[9];
  const float* att_w     = (const float*)d_in[10];
  const float* att_b     = (const float*)d_in[11];
  const float* fk_w      = (const float*)d_in[12];
  const float* fk_b      = (const float*)d_in[13];
  const float* bn_g      = (const float*)d_in[14];
  const float* bn_b      = (const float*)d_in[15];
  const float* pool_w    = (const float*)d_in[16];
  const float* pool_b    = (const float*)d_in[17];
  const float* head_w1   = (const float*)d_in[18];
  const float* head_b1   = (const float*)d_in[19];
  const float* e_w2 = (const float*)d_in[20]; const float* e_b2 = (const float*)d_in[21];
  const float* s_w2 = (const float*)d_in[22]; const float* s_b2 = (const float*)d_in[23];
  const float* c_w2 = (const float*)d_in[24]; const float* c_b2 = (const float*)d_in[25];
  const float* m_w2 = (const float*)d_in[26]; const float* m_b2 = (const float*)d_in[27];

  const int* src = ei;
  const int* dst = ei + EE;

  // workspace carve-up (256B aligned)
  char* ws = (char*)d_ws;
  size_t off = 0;
  auto carve = [&](size_t bytes) { char* p = ws + off; off = (off + bytes + 255) & ~(size_t)255; return p; };
  float*    hA     = (float*)   carve((size_t)NN * HID * 4);
  float*    hB     = (float*)   carve((size_t)NN * HID * 4);
  __bf16*   hbf    = (__bf16*)  carve((size_t)NN * HID * 2);
  __bf16*   linwt  = (__bf16*)  carve((size_t)DEPTH * WC * KW * 2);
  __bf16*   fkwt   = (__bf16*)  carve((size_t)DEPTH * HID * FKK * 2);
  float*    stats  = (float*)   carve(256 * 4);
  unsigned* redmax = (unsigned*)carve(64);
  float*    redsum = (float*)   carve(64);
  float*    logits = (float*)   carve((size_t)NN * 4);
  float*    pooled = (float*)   carve((size_t)BB * HID * 4);
  float*    comb   = (float*)   carve((size_t)BB * 2 * HID * 4);
  (void)ws_size; (void)in_sizes; (void)n_in; (void)out_size;

  // weight prep (bf16, transposed, K-padded, att_w folded in as extra column)
  k_prep_lin<<<(DEPTH * WC * KW + 255) / 256, 256, 0, stream>>>(lin_w, att_w, linwt);
  k_prep_fk <<<(DEPTH * HID * FKK + 255) / 256, 256, 0, stream>>>(fk_w, fkwt);

  // node embedding
  k_embed<<<(NN * HID + 255) / 256, 256, 0, stream>>>(x, node_w, node_b, hA, hbf);

  for (int l = 0; l < DEPTH; ++l) {
    k_zero<<<(NN * HID + 255) / 256, 256, 0, stream>>>(hB, NN * HID);
    k_edge<<<(EE + TE - 1) / TE, 128, 0, stream>>>(
        hbf, src, dst, eattr,
        linwt + (size_t)l * WC * KW, lin_b + l * HID, att_b + l, hB);
    k_fk<<<(NN + 15) / 16, 128, 0, stream>>>(
        hA, fkwt + (size_t)l * HID * FKK, fk_b + l * HID, hB);
    k_zero<<<1, 256, 0, stream>>>(stats, 256);
    k_bn_stats<<<(NN + 31) / 32, 256, 0, stream>>>(hB, stats);
    k_bn_apply<<<(NN * HID + 255) / 256, 256, 0, stream>>>(
        hB, stats, bn_g + l * HID, bn_b + l * HID, hA, hbf);
  }

  // softmax attention pool over all nodes
  k_init_red<<<1, 1, 0, stream>>>(redmax, redsum);
  k_zero<<<(BB * HID + 255) / 256, 256, 0, stream>>>(pooled, BB * HID);
  k_logits<<<(NN + 255) / 256, 256, 0, stream>>>(hA, pool_w, pool_b, logits, redmax);
  k_expsum<<<(NN + 255) / 256, 256, 0, stream>>>(logits, redmax, redsum);
  k_pool  <<<(NN * HID + 255) / 256, 256, 0, stream>>>(hA, logits, redsum, batch, pooled);
  k_comb  <<<(BB * 2 * HID + 255) / 256, 256, 0, stream>>>(pooled, sg_table, sgid, comb);

  // heads -> d_out (energy | stability | crystal | material, flat)
  k_heads<<<BB, 128, 0, stream>>>(comb, head_w1, head_b1,
                                  e_w2, e_b2, s_w2, s_b2, c_w2, c_b2, m_w2, m_b2,
                                  (float*)d_out);
}
